// RNN_64415919506221
// MI455X (gfx1250) — matrix operand order repaired and checked by reading, not compiled, not run on hardware
//
#include <hip/hip_runtime.h>
#include <hip/hip_bf16.h>

#define VOCAB 30522
#define EMB   64
#define HID   128
#define BATCH 512
#define SEQ   512

typedef _Float16 v16h __attribute__((ext_vector_type(16)));
typedef _Float16 v8h  __attribute__((ext_vector_type(8)));
typedef float    v8f  __attribute__((ext_vector_type(8)));

// LDS row strides (in halfs), padded so 16-lane fragment loads hit distinct banks
#define WRS 136   // W_hh rows: 272B = 68 dwords -> bank (4n+c)%64, distinct per lane
#define IRS 72    // W_ih rows: 144B = 36 dwords -> distinct
#define XRS 72    // xe tile rows
#define HRS 136   // h tile rows

// Build a 16-half fragment from two 16B-aligned LDS chunks (2x ds_load_b128).
__device__ __forceinline__ v16h ld_frag(const _Float16* p0, const _Float16* p1) {
  v8h lo = *(const v8h*)p0;
  v8h hi = *(const v8h*)p1;
  return __builtin_shufflevector(lo, hi, 0,1,2,3,4,5,6,7,8,9,10,11,12,13,14,15);
}

// Branchless tanh: native v_tanh_f32 on gfx1250, else fast-exp identity.
__device__ __forceinline__ float fast_tanh(float v) {
#if __has_builtin(__builtin_amdgcn_tanhf)
  return __builtin_amdgcn_tanhf(v);
#else
  float e = __expf(2.0f * v);                       // v_mul + v_exp_f32
  return 1.0f - 2.0f * __builtin_amdgcn_rcpf(e + 1.0f);
#endif
}

__global__ __launch_bounds__(256, 1) void rnn_fused(
    const int* __restrict__ x, const float* __restrict__ emb,
    const float* __restrict__ W_ih, const float* __restrict__ W_hh,
    const float* __restrict__ b_ih, const float* __restrict__ b_hh,
    const float* __restrict__ W_fc, const float* __restrict__ b_fc,
    float* __restrict__ out)
{
  __shared__ __align__(16) _Float16 sWhh[HID * WRS]; // 34816 B
  __shared__ __align__(16) _Float16 sWih[HID * IRS]; // 18432 B
  __shared__ __align__(16) _Float16 sXe[16 * XRS];   //  2304 B
  __shared__ __align__(16) _Float16 sH [16 * HRS];   //  4352 B  (total ~58.5 KB)

  const int t    = threadIdx.x;
  const int lane = t & 31;          // wave32
  const int wave = t >> 5;          // 8 waves; wave w owns HID columns [16w,16w+16)
  const int g    = lane >> 4;       // K-half group
  const int ln   = lane & 15;       // row (A) / column (B,C,D) within tile
  const int n0   = wave * 16;
  const int b0   = blockIdx.x * 16; // 16 batch rows per workgroup

  // One-time: convert weights fp32 -> f16 into padded LDS
  for (int i = t; i < HID * HID; i += 256) { int n = i >> 7, k = i & 127; sWhh[n * WRS + k] = (_Float16)W_hh[i]; }
  for (int i = t; i < HID * EMB; i += 256) { int n = i >> 6, k = i & 63;  sWih[n * IRS + k] = (_Float16)W_ih[i]; }
  for (int i = t; i < 16 * HRS;  i += 256) sH[i] = (_Float16)0.f;  // h0 = 0
  __syncthreads();

  // Hoist loop-invariant B-matrix (weight) fragments into registers (48 VGPRs):
  // B(32x16 f16): lanes0-15 hold K=k0..k0+7,k0+16..k0+23 of column N=ln, lanes16-31 K=k0+8..k0+15,k0+24..k0+31.
  // Column n of B equals row n of W (since xp = A @ W^T), so rows are contiguous.
  v16h Bih[2], Bhh[4];
  {
    const _Float16* wr = &sWih[(n0 + ln) * IRS];
#pragma unroll
    for (int kk = 0; kk < 2; ++kk) { const _Float16* p = wr + kk * 32 + g * 8; Bih[kk] = ld_frag(p, p + 16); }
    const _Float16* hr = &sWhh[(n0 + ln) * WRS];
#pragma unroll
    for (int kk = 0; kk < 4; ++kk) { const _Float16* p = hr + kk * 32 + g * 8; Bhh[kk] = ld_frag(p, p + 16); }
  }

  const int   hcol = n0 + ln;                 // D-matrix column this lane owns
  const float bias = b_ih[hcol] + b_hh[hcol]; // uniform across C rows

  // Embedding gather assignment: thread t loads float4 chunk q of batch row r.
  // Software pipeline: cur4 = chunk for step s (in regs), tokn = token for s+1.
  const int r = t >> 4;
  const int q = t & 15;
  const int* xrow = x + (size_t)(b0 + r) * SEQ;

  int tok0 = xrow[0];
  int tokn = xrow[1];
  float4 cur4 = reinterpret_cast<const float4*>(emb + (size_t)tok0 * EMB)[q];

  for (int s = 0; s < SEQ; ++s) {
    // Commit step-s embedding chunk (already in registers) to the shared A tile
    {
      union { _Float16 h[4]; unsigned long long u; } pk;
      pk.h[0] = (_Float16)cur4.x; pk.h[1] = (_Float16)cur4.y;
      pk.h[2] = (_Float16)cur4.z; pk.h[3] = (_Float16)cur4.w;
      *(unsigned long long*)&sXe[r * XRS + q * 4] = pk.u;
    }
    __syncthreads();

    // Issue next step's gather + token fetch now; consumed next iteration, so
    // the whole WMMA chain + tanh + barrier hides the L2 latency. Clamped
    // indices keep this branch-free (last-iteration loads are harmless).
    float4 nxt4 = reinterpret_cast<const float4*>(emb + (size_t)tokn * EMB)[q];
    int tokn2 = xrow[(s + 2 < SEQ) ? (s + 2) : (SEQ - 1)];

    // Load all 6 A-fragments first so the 12 ds_load_b128 pipeline together
    v16h A[6];
#pragma unroll
    for (int kk = 0; kk < 2; ++kk) {
      const _Float16* p = &sXe[ln * XRS + kk * 32 + g * 8];
      A[kk] = ld_frag(p, p + 16);
    }
#pragma unroll
    for (int kk = 0; kk < 4; ++kk) {
      const _Float16* p = &sH[ln * HRS + kk * 32 + g * 8];
      A[2 + kk] = ld_frag(p, p + 16);
    }

    v8f acc = {bias, bias, bias, bias, bias, bias, bias, bias};
    // Input projection: xe(16x64) @ W_ih^T  (K=64, 2 WMMAs)
#pragma unroll
    for (int kk = 0; kk < 2; ++kk)
      acc = __builtin_amdgcn_wmma_f32_16x16x32_f16(false, A[kk], false, Bih[kk], (short)0, acc, false, false);
    // Recurrent: h(16x128) @ W_hh^T  (K=128, 4 WMMAs)
#pragma unroll
    for (int kk = 0; kk < 4; ++kk)
      acc = __builtin_amdgcn_wmma_f32_16x16x32_f16(false, A[2 + kk], false, Bhh[kk], (short)0, acc, false, false);

    float hn[8];
#pragma unroll
    for (int i = 0; i < 8; ++i) hn[i] = fast_tanh(acc[i]);

    __syncthreads();  // all waves done reading sH/sXe for this step
    // D layout: VGPR i -> row i (lanes0-15) / row i+8 (lanes16-31), col = ln
#pragma unroll
    for (int i = 0; i < 8; ++i)
      sH[(i + g * 8) * HRS + hcol] = (_Float16)hn[i];

    cur4 = nxt4;
    tokn = tokn2;
  }
  __syncthreads();

  // Final FC (128-dot) + sigmoid: one thread per batch row; negligible cost
  if (t < 16) {
    float sum = b_fc[0];
#pragma unroll 4
    for (int k = 0; k < HID; ++k) sum += (float)sH[t * HRS + k] * W_fc[k];
    out[b0 + t] = __builtin_amdgcn_rcpf(1.0f + __expf(-sum));
  }
}

extern "C" void kernel_launch(void* const* d_in, const int* in_sizes, int n_in,
                              void* d_out, int out_size, void* d_ws, size_t ws_size,
                              hipStream_t stream) {
  (void)in_sizes; (void)n_in; (void)d_ws; (void)ws_size; (void)out_size;
  const int*   x    = (const int*)  d_in[0];
  const float* emb  = (const float*)d_in[1];
  const float* W_ih = (const float*)d_in[2];
  const float* W_hh = (const float*)d_in[3];
  const float* b_ih = (const float*)d_in[4];
  const float* b_hh = (const float*)d_in[5];
  const float* W_fc = (const float*)d_in[6];
  const float* b_fc = (const float*)d_in[7];
  rnn_fused<<<BATCH / 16, 256, 0, stream>>>(x, emb, W_ih, W_hh, b_ih, b_hh,
                                            W_fc, b_fc, (float*)d_out);
}
